// CloudCrop_23682449670885
// MI455X (gfx1250) — matrix operand order repaired and checked by reading, not compiled, not run on hardware
//
#include <hip/hip_runtime.h>

// ---------------------------------------------------------------------------
// CloudCrop for MI455X (gfx1250, wave32, WMMA).
// Stage A: ballot/compact grouping (exact first-64-by-index semantics).
// Stage B: fold BN into weights, pre-pack v_wmma A-fragments (ISA 7.12.2).
// Stage C: fully fused 3-layer MLP per 64-sample group using
//          v_wmma_f32_16x16x32_f16, + max-over-samples reduction.
// ---------------------------------------------------------------------------

typedef __attribute__((ext_vector_type(16))) _Float16 v16h;
typedef __attribute__((ext_vector_type(8)))  float    v8f;

#define B_      2
#define S_      1024
#define N_      16384
#define D_      4
#define NS_     64                 // NSAMPLE
#define GROUPS_ (B_ * S_ * D_)     // 8192
#define RADIUS_ 0.05f
#define HMIN_   (-0.02f)
#define BN_EPS_ 1e-5f

// ---------------------------------------------------------------------------
// Stage A: grouping. One wave (32 lanes) per seed; scan N points in index
// order, ballot-compact first 64 valid per hmax, pad with first-valid (or
// point 0's d_rot if none). Output x0[group][slot][4] f16 (ch0..2, pad).
// ---------------------------------------------------------------------------
__device__ __forceinline__ void store_x0(_Float16* dst, float a, float b, float c) {
    union { _Float16 h[4]; uint2 u; } t;
    t.h[0] = (_Float16)a; t.h[1] = (_Float16)b; t.h[2] = (_Float16)c; t.h[3] = (_Float16)0.0f;
    *(uint2*)dst = t.u;
}

__global__ __launch_bounds__(32) void group_kernel(
    const float* __restrict__ seed, const float* __restrict__ pc,
    const float* __restrict__ rot, _Float16* __restrict__ x0)
{
    const int bs   = blockIdx.x;           // b*S + s
    const int b    = bs / S_;
    const int lane = threadIdx.x;

    const float* sp = seed + (size_t)bs * 3;
    const float* rp = rot  + (size_t)bs * 9;
    const float sx = sp[0], sy = sp[1], sz = sp[2];
    const float r00 = rp[0], r01 = rp[1], r02 = rp[2];
    const float r10 = rp[3], r11 = rp[4], r12 = rp[5];
    const float r20 = rp[6], r21 = rp[7], r22 = rp[8];
    const float hmax[4] = {0.01f, 0.02f, 0.03f, 0.04f};
    const float R2 = RADIUS_ * RADIUS_;

    unsigned cnt[4] = {0u, 0u, 0u, 0u};
    bool  has[4] = {false, false, false, false};
    float fv0[4], fv1[4], fv2[4];
    float p00 = 0.f, p01 = 0.f, p02 = 0.f;    // d_rot of point index 0
    const unsigned long long ltmask = (1ull << lane) - 1ull;

    for (int it = 0; it < N_ / 32; ++it) {
        const int p = it * 32 + lane;
        const float* pp = pc + (size_t)(b * N_ + p) * 3;
        const float dx = pp[0] - sx, dy = pp[1] - sy, dz = pp[2] - sz;
        const float e0 = dx * r00 + dy * r10 + dz * r20;
        const float e1 = dx * r01 + dy * r11 + dz * r21;
        const float e2 = dx * r02 + dy * r12 + dz * r22;
        if (it == 0) {
            p00 = __shfl(e0, 0, 32); p01 = __shfl(e1, 0, 32); p02 = __shfl(e2, 0, 32);
        }
        const float r2 = e1 * e1 + e2 * e2;
        const bool base = (r2 < R2) && (e0 > HMIN_);

        #pragma unroll
        for (int gi = 0; gi < 4; ++gi) {
            if (cnt[gi] >= (unsigned)NS_) continue;          // wave-uniform
            const bool m = base && (e0 < hmax[gi]);
            const unsigned long long bal = __ballot((int)m);
            if (bal == 0ull) continue;                       // wave-uniform
            if (!has[gi]) {
                const int src = __ffsll(bal) - 1;
                fv0[gi] = __shfl(e0, src, 32);
                fv1[gi] = __shfl(e1, src, 32);
                fv2[gi] = __shfl(e2, src, 32);
                has[gi] = true;
            }
            const unsigned rank = (unsigned)__popcll(bal & ltmask);
            const unsigned slot = cnt[gi] + rank;
            if (m && slot < (unsigned)NS_) {
                const int g = bs * D_ + gi;
                store_x0(x0 + ((size_t)g * NS_ + slot) * 4, e0, e1, e2);
            }
            cnt[gi] += (unsigned)__popcll(bal);
        }
        if (cnt[0] >= NS_ && cnt[1] >= NS_ && cnt[2] >= NS_ && cnt[3] >= NS_) break;
    }

    #pragma unroll
    for (int gi = 0; gi < 4; ++gi) {
        const float f0 = has[gi] ? fv0[gi] : p00;
        const float f1 = has[gi] ? fv1[gi] : p01;
        const float f2 = has[gi] ? fv2[gi] : p02;
        const unsigned start = cnt[gi] > (unsigned)NS_ ? (unsigned)NS_ : cnt[gi];
        const int g = bs * D_ + gi;
        for (unsigned slot = start + (unsigned)lane; slot < (unsigned)NS_; slot += 32u)
            store_x0(x0 + ((size_t)g * NS_ + slot) * 4, f0, f1, f2);
    }
}

// ---------------------------------------------------------------------------
// Stage B: fold BN, pre-pack WMMA A-fragments.
// A-fragment (16x32 f16, this code's contiguous k order): lane l, elem e ->
//   M = l&15,  K = e + 16*(l>>4)
// ---------------------------------------------------------------------------
__device__ __forceinline__ void pack_layer(
    int tid, int stride, int MT, int KT, int cin, int cout,
    const float* __restrict__ w, const float* __restrict__ g,
    const float* __restrict__ v, _Float16* __restrict__ wA)
{
    const int total = MT * KT * 512;                    // frags * 32 lanes * 16 elems
    for (int i = tid; i < total; i += stride) {
        const int frag = i >> 9;
        const int rem  = i & 511;
        const int lane = rem >> 4;
        const int e    = rem & 15;
        const int mt   = frag / KT, kt = frag % KT;
        const int m    = mt * 16 + (lane & 15);
        const int k    = kt * 32 + (e + ((lane >> 4) << 4));
        float val = 0.f;
        if (k < cin && m < cout) {
            const float sc = g[m] * rsqrtf(v[m] + BN_EPS_);
            val = w[m * cin + k] * sc;
        }
        wA[i] = (_Float16)val;
    }
}

__global__ void prep_kernel(
    const float* w1, const float* b1, const float* g1, const float* be1, const float* m1, const float* v1,
    const float* w2, const float* b2, const float* g2, const float* be2, const float* m2, const float* v2,
    const float* w3, const float* b3, const float* g3, const float* be3, const float* m3, const float* v3,
    _Float16* w1A, _Float16* w2A, _Float16* w3A,
    float* bias1, float* bias2, float* bias3)
{
    const int tid    = blockIdx.x * blockDim.x + threadIdx.x;
    const int stride = gridDim.x * blockDim.x;
    pack_layer(tid, stride,  4, 1,   3,  64, w1, g1, v1, w1A);
    pack_layer(tid, stride,  8, 2,  64, 128, w2, g2, v2, w2A);
    pack_layer(tid, stride, 16, 4, 128, 256, w3, g3, v3, w3A);
    for (int o = tid; o <  64; o += stride) { float sc = g1[o] * rsqrtf(v1[o] + BN_EPS_); bias1[o] = (b1[o] - m1[o]) * sc + be1[o]; }
    for (int o = tid; o < 128; o += stride) { float sc = g2[o] * rsqrtf(v2[o] + BN_EPS_); bias2[o] = (b2[o] - m2[o]) * sc + be2[o]; }
    for (int o = tid; o < 256; o += stride) { float sc = g3[o] * rsqrtf(v3[o] + BN_EPS_); bias3[o] = (b3[o] - m3[o]) * sc + be3[o]; }
}

// ---------------------------------------------------------------------------
// Stage C: fused 3-layer MLP + max over 64 samples. One block per group,
// 4 waves; wave w owns columns [16w,16w+16). LDS activations stored [col][ch]
// so B-fragments (lane=col, K = e + 16*(lane>>4)) are contiguous 32B loads.
// ---------------------------------------------------------------------------
__device__ __forceinline__ v16h wmma_mm(v16h a, v16h b, v8f& c) {
    c = __builtin_amdgcn_wmma_f32_16x16x32_f16(false, a, false, b, (short)0, c, false, false);
    return a;
}

__global__ __launch_bounds__(128) void mlp_kernel(
    const _Float16* __restrict__ x0,
    const _Float16* __restrict__ w1A, const _Float16* __restrict__ w2A, const _Float16* __restrict__ w3A,
    const float* __restrict__ bias1, const float* __restrict__ bias2, const float* __restrict__ bias3,
    float* __restrict__ out)
{
    __shared__ __align__(32) _Float16 xb1[64 * 64];    //  8 KB, layer-1 output
    __shared__ __align__(32) _Float16 xb2[64 * 128];   // 16 KB, layer-2 output
    __shared__ float partial[4 * 256];                 //  4 KB, per-wave col-maxes

    const int g    = blockIdx.x;
    const int lane = threadIdx.x & 31;
    const int wave = threadIdx.x >> 5;
    const int l15  = lane & 15, lh = lane >> 4;
    const int n0   = wave * 16;
    const int col  = n0 + l15;

    // ---- layer-1 B fragment: cin=3 (K<3 lives in lanes<16, elems 0..2) ----
    v16h b0 = {};
    if (lh == 0) {
        const _Float16* p = x0 + ((size_t)g * NS_ + col) * 4;
        b0[0] = p[0]; b0[1] = p[1]; b0[2] = p[2];
    }

    // ---- layer 1: 3 -> 64 ----
    #pragma unroll
    for (int mt = 0; mt < 4; ++mt) {
        v16h a = *(const v16h*)(w1A + (size_t)(mt * 32 + lane) * 16);
        v8f  c = {};
        wmma_mm(a, b0, c);
        #pragma unroll
        for (int r = 0; r < 8; r += 2) {
            const int ch = mt * 16 + lh * 8 + r;
            float y0 = c[r]     + bias1[ch];     y0 = y0 > 0.f ? y0 : 0.f;
            float y1 = c[r + 1] + bias1[ch + 1]; y1 = y1 > 0.f ? y1 : 0.f;
            union { _Float16 h[2]; unsigned u; } t;
            t.h[0] = (_Float16)y0; t.h[1] = (_Float16)y1;
            *(unsigned*)&xb1[col * 64 + ch] = t.u;
        }
    }
    __syncthreads();

    // ---- layer 2: 64 -> 128 ----
    #pragma unroll
    for (int mt = 0; mt < 8; ++mt) {
        v8f c = {};
        #pragma unroll
        for (int kt = 0; kt < 2; ++kt) {
            v16h a = *(const v16h*)(w2A + (size_t)((mt * 2 + kt) * 32 + lane) * 16);
            v16h bf = *(const v16h*)(&xb1[col * 64 + kt * 32 + lh * 16]);
            wmma_mm(a, bf, c);
        }
        #pragma unroll
        for (int r = 0; r < 8; r += 2) {
            const int ch = mt * 16 + lh * 8 + r;
            float y0 = c[r]     + bias2[ch];     y0 = y0 > 0.f ? y0 : 0.f;
            float y1 = c[r + 1] + bias2[ch + 1]; y1 = y1 > 0.f ? y1 : 0.f;
            union { _Float16 h[2]; unsigned u; } t;
            t.h[0] = (_Float16)y0; t.h[1] = (_Float16)y1;
            *(unsigned*)&xb2[col * 128 + ch] = t.u;
        }
    }
    __syncthreads();

    // ---- layer 3: 128 -> 256, fused max over this wave's 16 columns ----
    #pragma unroll
    for (int mt = 0; mt < 16; ++mt) {
        v8f c = {};
        #pragma unroll
        for (int kt = 0; kt < 4; ++kt) {
            v16h a = *(const v16h*)(w3A + (size_t)((mt * 4 + kt) * 32 + lane) * 16);
            v16h bf = *(const v16h*)(&xb2[col * 128 + kt * 32 + lh * 16]);
            wmma_mm(a, bf, c);
        }
        #pragma unroll
        for (int r = 0; r < 8; ++r) {
            const int ch = mt * 16 + lh * 8 + r;
            float y = c[r] + bias3[ch]; y = y > 0.f ? y : 0.f;
            y = fmaxf(y, __shfl_xor(y, 1, 32));      // butterfly max over the
            y = fmaxf(y, __shfl_xor(y, 2, 32));      // 16 columns of this half
            y = fmaxf(y, __shfl_xor(y, 4, 32));
            y = fmaxf(y, __shfl_xor(y, 8, 32));
            if (l15 == 0) partial[wave * 256 + ch] = y;
        }
    }
    __syncthreads();

    // ---- combine 4 waves, write out[b][ch][s][d] ----
    const int b   = g >> 12;          // g / (S*D)
    const int rem = g & 4095;
    const int s   = rem >> 2;
    const int d   = rem & 3;
    #pragma unroll
    for (int i = 0; i < 2; ++i) {
        const int ch = threadIdx.x * 2 + i;
        const float y = fmaxf(fmaxf(partial[ch], partial[256 + ch]),
                              fmaxf(partial[512 + ch], partial[768 + ch]));
        out[(size_t)(((b * 256 + ch) * S_ + s) * D_ + d)] = y;
    }
}

// ---------------------------------------------------------------------------
extern "C" void kernel_launch(void* const* d_in, const int* in_sizes, int n_in,
                              void* d_out, int out_size, void* d_ws, size_t ws_size,
                              hipStream_t stream)
{
    const float* seed = (const float*)d_in[0];
    const float* pc   = (const float*)d_in[1];
    const float* rot  = (const float*)d_in[2];
    const float* w1 = (const float*)d_in[3],  *b1 = (const float*)d_in[4],
               * g1 = (const float*)d_in[5],  *be1 = (const float*)d_in[6],
               * m1 = (const float*)d_in[7],  *v1 = (const float*)d_in[8];
    const float* w2 = (const float*)d_in[9],  *b2 = (const float*)d_in[10],
               * g2 = (const float*)d_in[11], *be2 = (const float*)d_in[12],
               * m2 = (const float*)d_in[13], *v2 = (const float*)d_in[14];
    const float* w3 = (const float*)d_in[15], *b3 = (const float*)d_in[16],
               * g3 = (const float*)d_in[17], *be3 = (const float*)d_in[18],
               * m3 = (const float*)d_in[19], *v3 = (const float*)d_in[20];

    char* ws = (char*)d_ws;
    size_t off = 0;
    _Float16* x0  = (_Float16*)(ws + off); off += (size_t)GROUPS_ * NS_ * 4 * 2;  // 4 MB
    _Float16* w1A = (_Float16*)(ws + off); off += 4 * 512 * 2;                    // 4 KB
    _Float16* w2A = (_Float16*)(ws + off); off += 16 * 512 * 2;                   // 16 KB
    _Float16* w3A = (_Float16*)(ws + off); off += 64 * 512 * 2;                   // 64 KB
    float* bias1  = (float*)(ws + off);    off += 64 * 4;
    float* bias2  = (float*)(ws + off);    off += 128 * 4;
    float* bias3  = (float*)(ws + off);    off += 256 * 4;

    group_kernel<<<B_ * S_, 32, 0, stream>>>(seed, pc, rot, x0);
    prep_kernel<<<64, 256, 0, stream>>>(w1, b1, g1, be1, m1, v1,
                                        w2, b2, g2, be2, m2, v2,
                                        w3, b3, g3, be3, m3, v3,
                                        w1A, w2A, w3A, bias1, bias2, bias3);
    mlp_kernel<<<GROUPS_, 128, 0, stream>>>(x0, w1A, w2A, w3A,
                                            bias1, bias2, bias3, (float*)d_out);
}
